// TreeTransformer_13305808683027
// MI455X (gfx1250) — matrix operand order repaired and checked by reading, not compiled, not run on hardware
//
#include <hip/hip_runtime.h>
#include <hip/hip_bf16.h>
#include <math.h>

typedef __attribute__((ext_vector_type(16))) _Float16 v16h;
typedef __attribute__((ext_vector_type(8)))  _Float16 v8h;
typedef __attribute__((ext_vector_type(8)))  float    v8f;

namespace {
constexpr int SEQS = 1024;   // B*A
constexpr int NTOK = 121;
constexpr int NP   = 128;    // padded tokens
constexpr int HDIM = 128;
constexpr int H3   = 384;
constexpr int NH   = 4;
constexpr int HD   = 32;
constexpr int OUTD = 256;
constexpr int KOUT = NTOK * HDIM;   // 15488
constexpr float EPSF = 1e-5f;
}

// ---------------- WMMA fragment helpers (CDNA5 wave32, 16x16x32 f16) --------
// A fragment: A stored row-major [m][k] (lda halfs). ISA 7.12.2:
//  lanes 0-15: row=lane,    elems 0-7 -> K 0-7,  elems 8-15 -> K 16-23
//  lanes16-31: row=lane-16, elems 0-7 -> K 8-15, elems 8-15 -> K 24-31
__device__ __forceinline__ v16h load_a_frag(const _Float16* a, int lda) {
  int lane = threadIdx.x & 31;
  int row  = lane & 15;
  int ko   = (lane & 16) ? 8 : 0;
  const _Float16* p = a + row * lda + ko;
  v8h lo = *(const v8h*)(p);
  v8h hi = *(const v8h*)(p + 16);
  v16h r;
#pragma unroll
  for (int i = 0; i < 8; i++) { r[i] = lo[i]; r[8 + i] = hi[i]; }
  return r;
}
// B fragment: B stored TRANSPOSED as bt[n][k] (ldb halfs).
//  lanes 0-15: col=lane, elems 0-7 -> K 0-7, 8-15 -> K 16-23 ; lanes 16-31: K 8-15, K 24-31
__device__ __forceinline__ v16h load_b_frag(const _Float16* bt, int ldb) {
  int lane = threadIdx.x & 31;
  int col  = lane & 15;
  int kb   = (lane & 16) ? 8 : 0;
  const _Float16* p = bt + col * ldb + kb;
  v8h lo = *(const v8h*)(p);
  v8h hi = *(const v8h*)(p + 16);
  v16h r;
#pragma unroll
  for (int i = 0; i < 8; i++) { r[i] = lo[i]; r[8 + i] = hi[i]; }
  return r;
}
__device__ __forceinline__ v8f wmma_f16(v16h a, v16h b, v8f c) {
  return __builtin_amdgcn_wmma_f32_16x16x32_f16(false, a, false, b,
                                                (short)0, c, false, false);
}

// ---------------- K0: f32 -> f16 weight conversion ---------------------------
__global__ __launch_bounds__(256) void cvt_kernel(const float* __restrict__ in,
                                                  _Float16* __restrict__ out, int n) {
  int i = blockIdx.x * 256 + threadIdx.x;
  if (i < n) out[i] = (_Float16)in[i];
}

// ---------------- K1: embedding + tree positional encoding + permutation -----
__global__ __launch_bounds__(128) void embed_pe_perm_kernel(
    const float* __restrict__ forest, const int* __restrict__ perm,
    const float* __restrict__ w_in, const float* __restrict__ b_in,
    float* __restrict__ xf) {
  int blk = blockIdx.x;          // s*NP + no
  int s   = blk >> 7;
  int no  = blk & 127;
  int h   = threadIdx.x;
  float v = 0.0f;
  if (no < NTOK) {
    int n = perm[no];
    const float* f = forest + ((size_t)s * NTOK + n) * 12;
    const float* w = w_in + h * 12;
    float acc = b_in[h];
#pragma unroll
    for (int kf = 0; kf < 12; kf++) acc += f[kf] * w[kf];
    // depth (ternary tree, parent(c)=(c-1)/3), capped at 4 by DEPTH walk
    int c = n, d = 0;
#pragma unroll
    for (int i = 0; i < 4; i++) { if (c > 0) { d++; c = (c - 1) / 3; } }
    int cur = n;
#pragma unroll
    for (int i = 0; i < 4; i++) {
      if (d > 0) {
        int idx = (d - 1) * 3 + (cur - 1) % 3;
        if (idx == h) acc += 1.0f;
        cur = (cur - 1) / 3;
        d--;
      }
    }
    v = acc;
  }
  xf[((size_t)s * NP + no) * HDIM + h] = v;   // pad rows [121,128) = 0
}

// ---------------- K2: QKV projection (per sequence) --------------------------
// C(128x384) = x(128x128) @ Wqkv^T + b ; scatter into q[h][n][d], k[h][n][d],
// vt[h][d][n] (V transposed for the P@V B-operand). sect/head are uniform per
// 16-wide output tile -> resolve destination once per tile, branch-free.
__global__ __launch_bounds__(256) void qkv_kernel(
    const float* __restrict__ xf, const _Float16* __restrict__ wq,
    const float* __restrict__ bq, _Float16* __restrict__ q,
    _Float16* __restrict__ kk, _Float16* __restrict__ vt) {
  extern __shared__ char smem[];
  _Float16* As = (_Float16*)smem;      // [128][128]
  _Float16* Bs = As + NP * HDIM;       // [384][128]  ([n][k])
  int s = blockIdx.x, tid = threadIdx.x;
  const float* xs = xf + (size_t)s * NP * HDIM;
  for (int i = tid; i < NP * HDIM; i += 256) As[i] = (_Float16)xs[i];
  for (int i = tid; i < H3 * HDIM; i += 256) Bs[i] = wq[i];
  __syncthreads();
  int wave = tid >> 5, lane = tid & 31;
  for (int t = wave; t < 8 * 24; t += 8) {
    int tm = t / 24, tn = t % 24;
    int colg = tn * 16 + (lane & 15);
    float bias = bq[colg];
    v8f acc;
#pragma unroll
    for (int j = 0; j < 8; j++) acc[j] = bias;
#pragma unroll
    for (int kc = 0; kc < 4; kc++) {
      v16h a = load_a_frag(As + tm * 16 * HDIM + kc * 32, HDIM);
      v16h b = load_b_frag(Bs + tn * 16 * HDIM + kc * 32, HDIM);
      acc = wmma_f16(a, b, acc);
    }
    int rb   = (lane & 16) ? 8 : 0;
    int sect = tn >> 3;              // 0=q 1=k 2=v   (uniform per tile)
    int head = (tn >> 1) & 3;        // uniform per tile
    int dd   = ((tn & 1) << 4) + (lane & 15);  // 0..31 within head
    // branch-free destination select (pointer + stride cndmask)
    _Float16* base = (sect == 0) ? q : (sect == 1) ? kk : vt;
    size_t off0 = (sect == 2)
        ? ((((size_t)s * NH + head) * HD + dd) * NP)          // vt[h][d][*]
        : ((((size_t)s * NH + head) * NP) * HD + dd);         // q/k[h][*][d]
    int stride = (sect == 2) ? 1 : HD;
    _Float16* dst = base + off0 + (size_t)(tm * 16 + rb) * stride;
#pragma unroll
    for (int j = 0; j < 8; j++)
      dst[j * stride] = (_Float16)acc[j];
  }
}

// ---------------- K3: attention per (sequence, head) -------------------------
__global__ __launch_bounds__(256) void attn_kernel(
    const _Float16* __restrict__ q, const _Float16* __restrict__ kkey,
    const _Float16* __restrict__ vt, _Float16* __restrict__ o) {
  extern __shared__ char smem[];
  _Float16* Qs  = (_Float16*)smem;          // [128][32]
  _Float16* Ks  = Qs + NP * HD;             // [128][32]  ([key][d] == [n][k])
  _Float16* Vts = Ks + NP * HD;             // [32][128]  ([d][key] == [n][k])
  float*    Ss  = (float*)(Vts + HD * NP);  // [128][128]
  _Float16* Ps  = (_Float16*)(Ss + NP * NP);// [128][128]
  int sh = blockIdx.x;                      // s*4 + head
  int tid = threadIdx.x;
  const _Float16* qg = q    + (size_t)sh * NP * HD;
  const _Float16* kg = kkey + (size_t)sh * NP * HD;
  const _Float16* vg = vt   + (size_t)sh * HD * NP;
  for (int i = tid; i < NP * HD; i += 256) { Qs[i] = qg[i]; Ks[i] = kg[i]; Vts[i] = vg[i]; }
  __syncthreads();
  int wave = tid >> 5, lane = tid & 31;
  // S = (Q @ K^T) / sqrt(32) : one 16x16x32 WMMA per tile
  for (int t = wave; t < 64; t += 8) {
    int tm = t >> 3, tn = t & 7;
    v8f acc;
#pragma unroll
    for (int j = 0; j < 8; j++) acc[j] = 0.0f;
    v16h a = load_a_frag(Qs + tm * 16 * HD, HD);
    v16h b = load_b_frag(Ks + tn * 16 * HD, HD);
    acc = wmma_f16(a, b, acc);
    int col = tn * 16 + (lane & 15);
    int rb  = (lane & 16) ? 8 : 0;
    float mask = (col < NTOK) ? 0.0f : -1e30f;
#pragma unroll
    for (int j = 0; j < 8; j++)
      Ss[(tm * 16 + rb + j) * NP + col] = acc[j] * 0.17677669529663687f + mask;
  }
  __syncthreads();
  // row softmax over 121 valid keys
  if (tid < NP) {
    int r = tid;
    float m = -1e30f;
    for (int c2 = 0; c2 < NTOK; c2++) m = fmaxf(m, Ss[r * NP + c2]);
    float sum = 0.0f;
    for (int c2 = 0; c2 < NTOK; c2++) sum += __expf(Ss[r * NP + c2] - m);
    float inv = 1.0f / sum;
    for (int c2 = 0; c2 < NTOK; c2++)
      Ps[r * NP + c2] = (_Float16)(__expf(Ss[r * NP + c2] - m) * inv);
    for (int c2 = NTOK; c2 < NP; c2++) Ps[r * NP + c2] = (_Float16)0.0f;
  }
  __syncthreads();
  // O = P @ V : tiles (tm 0..7, tn 0..1), K loop over 128 keys
  for (int t = wave; t < 16; t += 8) {
    int tm = t >> 1, tn = t & 1;
    v8f acc;
#pragma unroll
    for (int j = 0; j < 8; j++) acc[j] = 0.0f;
#pragma unroll
    for (int kc = 0; kc < 4; kc++) {
      v16h a = load_a_frag(Ps  + tm * 16 * NP + kc * 32, NP);
      v16h b = load_b_frag(Vts + tn * 16 * NP + kc * 32, NP);
      acc = wmma_f16(a, b, acc);
    }
    int dcol = tn * 16 + (lane & 15);
    int head = sh & 3, s = sh >> 2;
    int rb = (lane & 16) ? 8 : 0;
    _Float16* dst = o + ((size_t)s * NP + tm * 16 + rb) * HDIM + head * HD + dcol;
#pragma unroll
    for (int j = 0; j < 8; j++)
      dst[j * HDIM] = (_Float16)acc[j];
  }
}

// ---------------- K4/K6: GEMM + bias + residual + LayerNorm ------------------
__global__ __launch_bounds__(256) void gemm_res_ln_kernel(
    const _Float16* __restrict__ ain, const _Float16* __restrict__ w,
    const float* __restrict__ bias, const float* __restrict__ g,
    const float* __restrict__ bln, float* __restrict__ xf) {
  extern __shared__ char smem[];
  _Float16* As = (_Float16*)smem;            // [128][128]
  _Float16* Bs = As + NP * HDIM;             // [128][128] ([n][k])
  float*    Cs = (float*)(Bs + HDIM * HDIM); // [128][128]
  int s = blockIdx.x, tid = threadIdx.x;
  const _Float16* ag = ain + (size_t)s * NP * HDIM;
  float* xs = xf + (size_t)s * NP * HDIM;
  for (int i = tid; i < NP * HDIM; i += 256) As[i] = ag[i];
  for (int i = tid; i < HDIM * HDIM; i += 256) Bs[i] = w[i];
  __syncthreads();
  int wave = tid >> 5, lane = tid & 31;
  for (int t = wave; t < 64; t += 8) {
    int tm = t >> 3, tn = t & 7;
    int col = tn * 16 + (lane & 15);
    float b0 = bias[col];
    v8f acc;
#pragma unroll
    for (int j = 0; j < 8; j++) acc[j] = b0;
#pragma unroll
    for (int kc = 0; kc < 4; kc++) {
      v16h a = load_a_frag(As + tm * 16 * HDIM + kc * 32, HDIM);
      v16h b = load_b_frag(Bs + tn * 16 * HDIM + kc * 32, HDIM);
      acc = wmma_f16(a, b, acc);
    }
    int rb = (lane & 16) ? 8 : 0;
#pragma unroll
    for (int j = 0; j < 8; j++) {
      int r = tm * 16 + rb + j;
      Cs[r * HDIM + col] = acc[j] + xs[r * HDIM + col];   // residual (f32)
    }
  }
  __syncthreads();
  if (tid < NP) {
    int r = tid;
    if (r < NTOK) {
      float mu = 0.0f;
      for (int c2 = 0; c2 < HDIM; c2++) mu += Cs[r * HDIM + c2];
      mu *= (1.0f / HDIM);
      float var = 0.0f;
      for (int c2 = 0; c2 < HDIM; c2++) {
        float dv = Cs[r * HDIM + c2] - mu;
        var += dv * dv;
      }
      var *= (1.0f / HDIM);
      float inv = rsqrtf(var + EPSF);
      for (int c2 = 0; c2 < HDIM; c2++)
        xs[r * HDIM + c2] = (Cs[r * HDIM + c2] - mu) * inv * g[c2] + bln[c2];
    } else {
      for (int c2 = 0; c2 < HDIM; c2++) xs[r * HDIM + c2] = 0.0f;  // keep pad zero
    }
  }
}

// ---------------- K5: FFN first GEMM + ReLU (f16 out) ------------------------
__global__ __launch_bounds__(256) void ffn1_kernel(
    const float* __restrict__ xf, const _Float16* __restrict__ w,
    const float* __restrict__ bias, _Float16* __restrict__ ff) {
  extern __shared__ char smem[];
  _Float16* As = (_Float16*)smem;     // [128][128]
  _Float16* Bs = As + NP * HDIM;      // [128][128]
  int s = blockIdx.x, tid = threadIdx.x;
  const float* xs = xf + (size_t)s * NP * HDIM;
  for (int i = tid; i < NP * HDIM; i += 256) As[i] = (_Float16)xs[i];
  for (int i = tid; i < HDIM * HDIM; i += 256) Bs[i] = w[i];
  __syncthreads();
  int wave = tid >> 5, lane = tid & 31;
  for (int t = wave; t < 64; t += 8) {
    int tm = t >> 3, tn = t & 7;
    int col = tn * 16 + (lane & 15);
    float b0 = bias[col];
    v8f acc;
#pragma unroll
    for (int j = 0; j < 8; j++) acc[j] = b0;
#pragma unroll
    for (int kc = 0; kc < 4; kc++) {
      v16h a = load_a_frag(As + tm * 16 * HDIM + kc * 32, HDIM);
      v16h b = load_b_frag(Bs + tn * 16 * HDIM + kc * 32, HDIM);
      acc = wmma_f16(a, b, acc);
    }
    int rb = (lane & 16) ? 8 : 0;
    _Float16* dst = ff + ((size_t)s * NP + tm * 16 + rb) * HDIM + col;
#pragma unroll
    for (int j = 0; j < 8; j++)
      dst[j * HDIM] = (_Float16)fmaxf(acc[j], 0.0f);
  }
}

// ---------------- K7: readout GEMM (K=15488) + final LayerNorm ---------------
__global__ __launch_bounds__(256) void out_ln_kernel(
    const float* __restrict__ xf, const _Float16* __restrict__ wout,
    const float* __restrict__ bout, const float* __restrict__ g,
    const float* __restrict__ bln, float* __restrict__ out) {
  extern __shared__ char smem[];
  _Float16* As = (_Float16*)smem;         // [16][64]
  _Float16* Bs = As + 16 * 64;            // [256][64] ([n][k])
  float*    Cs = (float*)(Bs + OUTD * 64);// [16][256]
  int sb = blockIdx.x * 16;               // 16 sequences per block
  int tid = threadIdx.x, wave = tid >> 5, lane = tid & 31;
  int tn0 = wave * 2;
  v8f acc0, acc1;
#pragma unroll
  for (int j = 0; j < 8; j++) { acc0[j] = 0.0f; acc1[j] = 0.0f; }
  for (int kb2 = 0; kb2 < KOUT / 64; kb2++) {     // 242 chunks of K=64
    // stage A: 16 rows x 64 (k = n*128+h, n = kb2/2, h = (kb2&1)*64 + kk)
    for (int i = tid; i < 16 * 64; i += 256) {
      int m = i >> 6, kk2 = i & 63;
      int n = kb2 >> 1, h0 = (kb2 & 1) * 64 + kk2;
      As[i] = (_Float16)xf[(((size_t)(sb + m)) * NP + n) * HDIM + h0];
    }
    // stage B: thread tid copies row tid's 64-halfs chunk (128B, v8h x8)
    {
      const _Float16* src = wout + (size_t)tid * KOUT + kb2 * 64;
      __builtin_prefetch(src + 64, 0, 1);          // global_prefetch next chunk
      v8h* dst = (v8h*)(Bs + tid * 64);
      const v8h* sp = (const v8h*)src;
#pragma unroll
      for (int u = 0; u < 8; u++) dst[u] = sp[u];
    }
    __syncthreads();
#pragma unroll
    for (int kc = 0; kc < 2; kc++) {
      v16h a  = load_a_frag(As + kc * 32, 64);
      v16h b0 = load_b_frag(Bs + (tn0 * 16) * 64 + kc * 32, 64);
      v16h b1 = load_b_frag(Bs + ((tn0 + 1) * 16) * 64 + kc * 32, 64);
      acc0 = wmma_f16(a, b0, acc0);
      acc1 = wmma_f16(a, b1, acc1);
    }
    __syncthreads();
  }
  int rb = (lane & 16) ? 8 : 0;
  {
    int col = tn0 * 16 + (lane & 15);
    float bb = bout[col];
#pragma unroll
    for (int j = 0; j < 8; j++) Cs[(rb + j) * OUTD + col] = acc0[j] + bb;
  }
  {
    int col = (tn0 + 1) * 16 + (lane & 15);
    float bb = bout[col];
#pragma unroll
    for (int j = 0; j < 8; j++) Cs[(rb + j) * OUTD + col] = acc1[j] + bb;
  }
  __syncthreads();
  if (tid < 16) {
    int m = tid;
    float mu = 0.0f;
    for (int c2 = 0; c2 < OUTD; c2++) mu += Cs[m * OUTD + c2];
    mu *= (1.0f / OUTD);
    float var = 0.0f;
    for (int c2 = 0; c2 < OUTD; c2++) {
      float dv = Cs[m * OUTD + c2] - mu;
      var += dv * dv;
    }
    var *= (1.0f / OUTD);
    float inv = rsqrtf(var + EPSF);
    for (int c2 = 0; c2 < OUTD; c2++)
      out[(size_t)(sb + m) * OUTD + c2] =
          (Cs[m * OUTD + c2] - mu) * inv * g[c2] + bln[c2];
  }
}

// ---------------- host side --------------------------------------------------
extern "C" void kernel_launch(void* const* d_in, const int* in_sizes, int n_in,
                              void* d_out, int out_size, void* d_ws, size_t ws_size,
                              hipStream_t stream) {
  (void)in_sizes; (void)n_in; (void)out_size; (void)ws_size;
  const float* forest = (const float*)d_in[0];
  const int*   perm   = (const int*)d_in[4];
  const float* w_in   = (const float*)d_in[5];
  const float* b_in   = (const float*)d_in[6];
  const float* wqkv   = (const float*)d_in[7];
  const float* bqkv   = (const float*)d_in[8];
  const float* wo     = (const float*)d_in[9];
  const float* bo     = (const float*)d_in[10];
  const float* ln1g   = (const float*)d_in[11];
  const float* ln1b   = (const float*)d_in[12];
  const float* w1     = (const float*)d_in[13];
  const float* b1     = (const float*)d_in[14];
  const float* w2     = (const float*)d_in[15];
  const float* b2     = (const float*)d_in[16];
  const float* ln2g   = (const float*)d_in[17];
  const float* ln2b   = (const float*)d_in[18];
  const float* w_out  = (const float*)d_in[19];
  const float* b_out  = (const float*)d_in[20];
  const float* lnf_g  = (const float*)d_in[21];
  const float* lnf_b  = (const float*)d_in[22];

  // scratch carving (256B aligned)
  char* ws = (char*)d_ws;
  size_t off = 0;
  auto carve = [&](size_t bytes) -> char* {
    char* p = ws + off;
    off = (off + bytes + 255) & ~(size_t)255;
    return p;
  };
  _Float16* wqkv16 = (_Float16*)carve((size_t)2 * H3 * HDIM * 2);
  _Float16* wo16   = (_Float16*)carve((size_t)2 * HDIM * HDIM * 2);
  _Float16* w116   = (_Float16*)carve((size_t)2 * HDIM * HDIM * 2);
  _Float16* w216   = (_Float16*)carve((size_t)2 * HDIM * HDIM * 2);
  _Float16* wout16 = (_Float16*)carve((size_t)OUTD * KOUT * 2);
  float*    xf     = (float*)   carve((size_t)SEQS * NP * HDIM * 4);
  _Float16* qbuf   = (_Float16*)carve((size_t)SEQS * NH * NP * HD * 2);
  _Float16* kbuf   = (_Float16*)carve((size_t)SEQS * NH * NP * HD * 2);
  _Float16* vtbuf  = (_Float16*)carve((size_t)SEQS * NH * HD * NP * 2);
  _Float16* obuf   = (_Float16*)carve((size_t)SEQS * NP * HDIM * 2);
  _Float16* ffbuf  = (_Float16*)carve((size_t)SEQS * NP * HDIM * 2);

  constexpr int SM_QKV = (NP * HDIM + H3 * HDIM) * 2;                       // 128 KB
  constexpr int SM_ATT = (NP * HD * 3) * 2 + NP * NP * 4 + NP * NP * 2;     // 120 KB
  constexpr int SM_GRL = (NP * HDIM * 2) * 2 + NP * HDIM * 4;               // 128 KB
  constexpr int SM_FF1 = (NP * HDIM * 2) * 2;                               //  64 KB
  constexpr int SM_OUT = (16 * 64 + OUTD * 64) * 2 + 16 * OUTD * 4;         //  50 KB

  (void)hipFuncSetAttribute(reinterpret_cast<const void*>(qkv_kernel),
                            hipFuncAttributeMaxDynamicSharedMemorySize, SM_QKV);
  (void)hipFuncSetAttribute(reinterpret_cast<const void*>(attn_kernel),
                            hipFuncAttributeMaxDynamicSharedMemorySize, SM_ATT);
  (void)hipFuncSetAttribute(reinterpret_cast<const void*>(gemm_res_ln_kernel),
                            hipFuncAttributeMaxDynamicSharedMemorySize, SM_GRL);
  (void)hipFuncSetAttribute(reinterpret_cast<const void*>(ffn1_kernel),
                            hipFuncAttributeMaxDynamicSharedMemorySize, SM_FF1);
  (void)hipFuncSetAttribute(reinterpret_cast<const void*>(out_ln_kernel),
                            hipFuncAttributeMaxDynamicSharedMemorySize, SM_OUT);

  // weight conversions
  auto cvt = [&](const float* src, _Float16* dst, int n) {
    cvt_kernel<<<(n + 255) / 256, 256, 0, stream>>>(src, dst, n);
  };
  cvt(wqkv,  wqkv16, 2 * H3 * HDIM);
  cvt(wo,    wo16,   2 * HDIM * HDIM);
  cvt(w1,    w116,   2 * HDIM * HDIM);
  cvt(w2,    w216,   2 * HDIM * HDIM);
  cvt(w_out, wout16, OUTD * KOUT);

  // embedding + PE + permutation -> xf (padded to 128 rows/seq)
  embed_pe_perm_kernel<<<SEQS * NP, 128, 0, stream>>>(forest, perm, w_in, b_in, xf);

  for (int L = 0; L < 2; L++) {
    qkv_kernel<<<SEQS, 256, SM_QKV, stream>>>(
        xf, wqkv16 + (size_t)L * H3 * HDIM, bqkv + L * H3, qbuf, kbuf, vtbuf);
    attn_kernel<<<SEQS * NH, 256, SM_ATT, stream>>>(qbuf, kbuf, vtbuf, obuf);
    gemm_res_ln_kernel<<<SEQS, 256, SM_GRL, stream>>>(
        obuf, wo16 + (size_t)L * HDIM * HDIM, bo + L * HDIM,
        ln1g + L * HDIM, ln1b + L * HDIM, xf);
    ffn1_kernel<<<SEQS, 256, SM_FF1, stream>>>(
        xf, w116 + (size_t)L * HDIM * HDIM, b1 + L * HDIM, ffbuf);
    gemm_res_ln_kernel<<<SEQS, 256, SM_GRL, stream>>>(
        ffbuf, w216 + (size_t)L * HDIM * HDIM, b2 + L * HDIM,
        ln2g + L * HDIM, ln2b + L * HDIM, xf);
  }

  out_ln_kernel<<<SEQS / 16, 256, SM_OUT, stream>>>(
      xf, wout16, b_out, lnf_g, lnf_b, (float*)d_out);
}
